// MiniBatchDiscrimination_35596688949267
// MI455X (gfx1250) — matrix operand order repaired and checked by reading, not compiled, not run on hardware
//
#include <hip/hip_runtime.h>
#include <hip/hip_bf16.h>

typedef __attribute__((ext_vector_type(16))) _Float16 v16h;
typedef __attribute__((ext_vector_type(8)))  _Float16 v8h;
typedef __attribute__((ext_vector_type(8)))  float    v8f;

#define B_SZ    256
#define IN_DIM  1024
#define NUM_K   100
#define DIM_K   50
#define N_REAL  (NUM_K * DIM_K)   // 5000
#define N_PAD   5120              // padded so every 16/64-wide tile is full
#define OUT_DIM (IN_DIM + NUM_K)  // 1124

// ---------------- f32 -> f16 conversion of x ----------------
__global__ void cvt_x_f16(const float* __restrict__ x, _Float16* __restrict__ x16, int n) {
    int i = blockIdx.x * blockDim.x + threadIdx.x;
    if (i < n) x16[i] = (_Float16)x[i];
}

// ---------------- transpose + convert weight: w[K][5000] -> wt[N_PAD][K] (f16, zero-padded)
__global__ void cvt_w_f16t(const float* __restrict__ w, _Float16* __restrict__ wt) {
    int idx = blockIdx.x * blockDim.x + threadIdx.x;     // idx = k*N_PAD + n (n fastest: coalesced reads)
    if (idx >= IN_DIM * N_PAD) return;
    int k = idx / N_PAD;
    int n = idx - k * N_PAD;
    float v = (n < N_REAL) ? w[(size_t)k * N_REAL + n] : 0.0f;
    wt[(size_t)n * IN_DIM + k] = (_Float16)v;
}

// A tile fragment: per-lane halves contiguous at +0..15
__device__ __forceinline__ v16h load_a_frag(const _Float16* p) {
    v8h lo = *(const v8h*)(p);
    v8h hi = *(const v8h*)(p + 8);
    return __builtin_shufflevector(lo, hi, 0,1,2,3,4,5,6,7,8,9,10,11,12,13,14,15);
}
// B tile fragment: per-lane halves contiguous at +0..15
__device__ __forceinline__ v16h load_b_frag(const _Float16* p) {
    v8h lo = *(const v8h*)(p);
    v8h hi = *(const v8h*)(p + 8);
    return __builtin_shufflevector(lo, hi, 0,1,2,3,4,5,6,7,8,9,10,11,12,13,14,15);
}

// ---------------- GEMM: act[256][N_PAD] = x16[256][1024] @ Wt^T ----------------
// One wave per 16(M) x 64(N) strip; A reused across 4 WMMA tiles per k-step.
// Manual ping-pong double buffering: two k-steps per iteration, no register
// rotation copies; loads of step i+1 overlap WMMAs of step i.
__global__ __launch_bounds__(32) void gemm_wmma_f16(
        const _Float16* __restrict__ A16,   // [256][1024] row-major
        const _Float16* __restrict__ BT16,  // [N_PAD][1024] row-major (W transposed)
        float* __restrict__ act) {          // [256][N_PAD]
    const int m0   = blockIdx.x * 16;
    const int n0   = blockIdx.y * 64;
    const int lane = threadIdx.x & 31;
    const int lrow = lane & 15;
    const int lhal = lane >> 4;

    const _Float16* aP = A16 + (size_t)(m0 + lrow) * IN_DIM + lhal * 16;
    const _Float16* bP = BT16 + (size_t)(n0 + lrow) * IN_DIM + lhal * 16;
    const size_t bstr = (size_t)16 * IN_DIM;   // next 16-col tile of B

    v8f c[4] = {v8f{}, v8f{}, v8f{}, v8f{}};

    // Prologue: buffer 0 <- k-step 0
    v16h a0 = load_a_frag(aP);
    v16h b0_0 = load_b_frag(bP);
    v16h b0_1 = load_b_frag(bP + bstr);
    v16h b0_2 = load_b_frag(bP + 2 * bstr);
    v16h b0_3 = load_b_frag(bP + 3 * bstr);
    v16h a1, b1_0, b1_1, b1_2, b1_3;

    int k0;
    for (k0 = 32; k0 + 32 < IN_DIM; k0 += 64) {
        // buffer 1 <- step at k0, while consuming buffer 0
        a1   = load_a_frag(aP + k0);
        b1_0 = load_b_frag(bP + k0);
        b1_1 = load_b_frag(bP + bstr + k0);
        b1_2 = load_b_frag(bP + 2 * bstr + k0);
        b1_3 = load_b_frag(bP + 3 * bstr + k0);
        c[0] = __builtin_amdgcn_wmma_f32_16x16x32_f16(false, a0, false, b0_0, (short)0, c[0], false, false);
        c[1] = __builtin_amdgcn_wmma_f32_16x16x32_f16(false, a0, false, b0_1, (short)0, c[1], false, false);
        c[2] = __builtin_amdgcn_wmma_f32_16x16x32_f16(false, a0, false, b0_2, (short)0, c[2], false, false);
        c[3] = __builtin_amdgcn_wmma_f32_16x16x32_f16(false, a0, false, b0_3, (short)0, c[3], false, false);

        // buffer 0 <- step at k0+32, while consuming buffer 1
        a0   = load_a_frag(aP + k0 + 32);
        b0_0 = load_b_frag(bP + k0 + 32);
        b0_1 = load_b_frag(bP + bstr + k0 + 32);
        b0_2 = load_b_frag(bP + 2 * bstr + k0 + 32);
        b0_3 = load_b_frag(bP + 3 * bstr + k0 + 32);
        c[0] = __builtin_amdgcn_wmma_f32_16x16x32_f16(false, a1, false, b1_0, (short)0, c[0], false, false);
        c[1] = __builtin_amdgcn_wmma_f32_16x16x32_f16(false, a1, false, b1_1, (short)0, c[1], false, false);
        c[2] = __builtin_amdgcn_wmma_f32_16x16x32_f16(false, a1, false, b1_2, (short)0, c[2], false, false);
        c[3] = __builtin_amdgcn_wmma_f32_16x16x32_f16(false, a1, false, b1_3, (short)0, c[3], false, false);
    }

    // Epilogue: cur holds step at k0-32... exit with buffer 0 = step (IN_DIM-64),
    // last step (IN_DIM-32) still to load.
    a1   = load_a_frag(aP + k0);
    b1_0 = load_b_frag(bP + k0);
    b1_1 = load_b_frag(bP + bstr + k0);
    b1_2 = load_b_frag(bP + 2 * bstr + k0);
    b1_3 = load_b_frag(bP + 3 * bstr + k0);
    c[0] = __builtin_amdgcn_wmma_f32_16x16x32_f16(false, a0, false, b0_0, (short)0, c[0], false, false);
    c[1] = __builtin_amdgcn_wmma_f32_16x16x32_f16(false, a0, false, b0_1, (short)0, c[1], false, false);
    c[2] = __builtin_amdgcn_wmma_f32_16x16x32_f16(false, a0, false, b0_2, (short)0, c[2], false, false);
    c[3] = __builtin_amdgcn_wmma_f32_16x16x32_f16(false, a0, false, b0_3, (short)0, c[3], false, false);
    c[0] = __builtin_amdgcn_wmma_f32_16x16x32_f16(false, a1, false, b1_0, (short)0, c[0], false, false);
    c[1] = __builtin_amdgcn_wmma_f32_16x16x32_f16(false, a1, false, b1_1, (short)0, c[1], false, false);
    c[2] = __builtin_amdgcn_wmma_f32_16x16x32_f16(false, a1, false, b1_2, (short)0, c[2], false, false);
    c[3] = __builtin_amdgcn_wmma_f32_16x16x32_f16(false, a1, false, b1_3, (short)0, c[3], false, false);

    // C layout: VGPR r -> M = r + lhal*8, N = lane&15
    const int rbase = lhal * 8;
    #pragma unroll
    for (int t = 0; t < 4; ++t) {
        #pragma unroll
        for (int r = 0; r < 8; ++r) {
            act[(size_t)(m0 + rbase + r) * N_PAD + n0 + t * 16 + lrow] = c[t][r];
        }
    }
}

// ---------------- minibatch features: one block per kernel k ----------------
__global__ __launch_bounds__(256) void minibatch_features(
        const float* __restrict__ act, float* __restrict__ out) {
    const int k = blockIdx.x;          // 0..99
    const int i = threadIdx.x;         // 0..255 (row / batch index)
    __shared__ float s[B_SZ * DIM_K];  // 51200 bytes
    float my[DIM_K];

    const float* row = act + (size_t)i * N_PAD + k * DIM_K;
    #pragma unroll
    for (int d = 0; d < DIM_K; ++d) {
        float v = row[d];
        my[d] = v;
        s[i * DIM_K + d] = v;
    }
    __syncthreads();

    float acc = 0.0f;
    for (int j = 0; j < B_SZ; ++j) {
        const float* sr = s + j * DIM_K;
        float l1 = 0.0f;
        #pragma unroll
        for (int d = 0; d < DIM_K; ++d) l1 += fabsf(my[d] - sr[d]);
        acc += __expf(-l1);
    }
    out[(size_t)i * OUT_DIM + IN_DIM + k] = acc;
}

// ---------------- copy x into out[:, :1024] ----------------
__global__ void copy_x(const float* __restrict__ x, float* __restrict__ out, int n) {
    int idx = blockIdx.x * blockDim.x + threadIdx.x;
    if (idx < n) {
        int i = idx >> 10;        // / 1024
        int c = idx & 1023;       // % 1024
        out[(size_t)i * OUT_DIM + c] = x[idx];
    }
}

extern "C" void kernel_launch(void* const* d_in, const int* in_sizes, int n_in,
                              void* d_out, int out_size, void* d_ws, size_t ws_size,
                              hipStream_t stream) {
    (void)in_sizes; (void)n_in; (void)out_size; (void)ws_size;
    const float* x = (const float*)d_in[0];   // [256][1024]
    const float* w = (const float*)d_in[1];   // [1024][5000]
    float* out = (float*)d_out;               // [256][1124]

    char* ws = (char*)d_ws;
    _Float16* x16 = (_Float16*)ws;                                     //   524,288 B
    _Float16* wt  = (_Float16*)(ws + 524288);                          // 10,485,760 B
    float*    act = (float*)(ws + 524288 + 10485760);                  //  5,242,880 B (total ~15.5 MB)

    cvt_x_f16<<<(B_SZ * IN_DIM + 255) / 256, 256, 0, stream>>>(x, x16, B_SZ * IN_DIM);
    cvt_w_f16t<<<(IN_DIM * N_PAD + 255) / 256, 256, 0, stream>>>(w, wt);

    dim3 grid(B_SZ / 16, N_PAD / 64);   // 16 x 80 waves
    gemm_wmma_f16<<<grid, 32, 0, stream>>>(x16, wt, act);

    minibatch_features<<<NUM_K, 256, 0, stream>>>(act, out);
    copy_x<<<(B_SZ * IN_DIM + 255) / 256, 256, 0, stream>>>(x, out, B_SZ * IN_DIM);
}
